// AttentionBlock_34488587387358
// MI455X (gfx1250) — matrix operand order repaired and checked by reading, not compiled, not run on hardware
//
#include <hip/hip_runtime.h>

#define B_ 2
#define C_ 512
#define L_ 4096
#define HEADS_ 8
#define CH_ 64
#define GROUPS_ 32
#define CPG_ 16
#define EPS_ 1e-6f

typedef __bf16 bf16;
typedef bf16  v16bf __attribute__((ext_vector_type(16)));
typedef float v8f   __attribute__((ext_vector_type(8)));

static __device__ __forceinline__ int lane_id() { return (int)(threadIdx.x & 31u); }

static __device__ __forceinline__ v8f zero_v8f() {
  v8f z;
#pragma unroll
  for (int r = 0; r < 8; ++r) z[r] = 0.0f;
  return z;
}

// ---- WMMA fragment loaders (CDNA5 wave32 16-bit layouts, CDNA5 ISA) ----
// A fragment (16x32, M x K): element (m,k) at base[m*sm + k*sk].
// With sk==1 each lane reads two contiguous 16-byte runs -> (global|ds)_load_b128.
static __device__ __forceinline__ v16bf load_a_frag(const bf16* base, long sm, long sk) {
  const int l   = lane_id();
  const long m  = (long)(l & 15);
  const long kb = (long)((l & 16) >> 1);   // 0 or 8
  v16bf a;
#pragma unroll
  for (int j = 0; j < 8; ++j) a[j]     = base[m * sm + (kb + j) * sk];
#pragma unroll
  for (int j = 0; j < 8; ++j) a[8 + j] = base[m * sm + (kb + 16 + j) * sk];
  return a;
}

// B fragment (32x16, K x N): element (k,n) at base[k*sk + n*sn].
// With sk==1 each lane reads two contiguous 16-byte runs -> two b128 loads.
static __device__ __forceinline__ v16bf load_b_frag(const bf16* base, long sk, long sn) {
  const int l   = lane_id();
  const long n  = (long)(l & 15);
  const long kb = (long)((l & 16) >> 1);   // 0 or 8
  v16bf b;
#pragma unroll
  for (int j = 0; j < 8; ++j) b[j]     = base[(kb + j) * sk + n * sn];
#pragma unroll
  for (int j = 0; j < 8; ++j) b[8 + j] = base[(kb + 16 + j) * sk + n * sn];
  return b;
}

static __device__ __forceinline__ v8f wmma_bf16(v16bf a, v16bf b, v8f c) {
  return __builtin_amdgcn_wmma_f32_16x16x32_bf16(false, a, false, b, (short)0, c, false, false);
}

// ---------------- GroupNorm statistics: one block per (b, group) ----------------
__global__ void gn_stats_kernel(const float* __restrict__ x, float* __restrict__ stats) {
  __shared__ float ss[256];
  __shared__ float sq[256];
  const int  bg   = blockIdx.x;                  // b*32 + g
  const long base = (long)bg * CPG_ * L_;
  float s = 0.0f, q = 0.0f;
  for (int i = threadIdx.x; i < CPG_ * L_; i += blockDim.x) {
    float v = x[base + i];
    s += v; q += v * v;
  }
  ss[threadIdx.x] = s; sq[threadIdx.x] = q;
  __syncthreads();
  for (int off = 128; off > 0; off >>= 1) {
    if ((int)threadIdx.x < off) {
      ss[threadIdx.x] += ss[threadIdx.x + off];
      sq[threadIdx.x] += sq[threadIdx.x + off];
    }
    __syncthreads();
  }
  if (threadIdx.x == 0) {
    const float n   = (float)(CPG_ * L_);
    const float mu  = ss[0] / n;
    const float var = sq[0] / n - mu * mu;
    stats[bg]      = mu;
    stats[64 + bg] = rsqrtf(var + EPS_);
  }
}

// ---------------- fp32 -> bf16 cast ----------------
__global__ void cast_bf16_kernel(const float* __restrict__ src, bf16* __restrict__ dst, int n) {
  for (int i = blockIdx.x * blockDim.x + threadIdx.x; i < n; i += gridDim.x * blockDim.x)
    dst[i] = (bf16)src[i];
}

// ---- normalize + affine -> bf16, stored TRANSPOSED as xnt[b, l, c] (c contiguous) ----
__global__ void gn_apply_kernel(const float* __restrict__ x, const float* __restrict__ gamma,
                                const float* __restrict__ beta, const float* __restrict__ stats,
                                bf16* __restrict__ xnt) {
  const long total = (long)B_ * C_ * L_;
  for (long i = blockIdx.x * (long)blockDim.x + threadIdx.x; i < total;
       i += (long)gridDim.x * blockDim.x) {
    const int lcol = (int)(i % L_);
    const int c    = (int)((i / L_) % C_);
    const int b    = (int)(i / ((long)C_ * L_));
    const int bg   = (int)(i / ((long)CPG_ * L_));   // b*32 + g
    const float mu = stats[bg];
    const float rs = stats[64 + bg];
    const float v  = (x[i] - mu) * rs * gamma[c] + beta[c];
    xnt[((long)b * L_ + lcol) * C_ + c] = (bf16)v;
  }
}

// ---------------- QKV GEMM (single-generation pipelined k-loop: no copies, no spills) --------
// Outputs scattered into WMMA-friendly layouts:
//   q -> qt[bh, t, c]   k -> kt[bh, s, c]   v -> vb[bh, c, s]
// A 64-row M-supertile lies entirely in one q/k/v segment of one head (192 = 3*64),
// so the scatter is wave-uniform: seg = mt%3, head = mt/3.
__global__ void gemm_qkv_kernel(const bf16* __restrict__ wq, const bf16* __restrict__ xnt,
                                const float* __restrict__ bqkv,
                                bf16* __restrict__ qt, bf16* __restrict__ kt,
                                bf16* __restrict__ vb) {
  const int wid = blockIdx.x * (blockDim.x >> 5) + (threadIdx.x >> 5);
  const int NT = L_ / 16;           // 256
  const int MT = (3 * C_) / 64;     // 24
  const int nt = wid % NT;
  const int mt = (wid / NT) % MT;
  const int b  = wid / (NT * MT);
  const int l  = lane_id();
  const int n  = l & 15;
  const int mb = (l & 16) >> 1;

  v8f acc[4];
#pragma unroll
  for (int i = 0; i < 4; ++i) acc[i] = zero_v8f();

  const bf16* Bbase = xnt + ((long)b * L_ + nt * 16) * C_;
  const bf16* Abase = wq + (long)mt * 64 * C_;

  // one fragment generation live; WMMAs consume, then the same registers are refilled
  v16bf B0 = load_b_frag(Bbase, 1, C_);
  v16bf A0 = load_a_frag(Abase,           C_, 1);
  v16bf A1 = load_a_frag(Abase + 16 * C_, C_, 1);
  v16bf A2 = load_a_frag(Abase + 32 * C_, C_, 1);
  v16bf A3 = load_a_frag(Abase + 48 * C_, C_, 1);
  for (int kk = 0; kk < C_ - 32; kk += 32) {
    acc[0] = wmma_bf16(A0, B0, acc[0]);
    acc[1] = wmma_bf16(A1, B0, acc[1]);
    acc[2] = wmma_bf16(A2, B0, acc[2]);
    acc[3] = wmma_bf16(A3, B0, acc[3]);
    B0 = load_b_frag(Bbase + kk + 32, 1, C_);
    A0 = load_a_frag(Abase + kk + 32,           C_, 1);
    A1 = load_a_frag(Abase + 16 * C_ + kk + 32, C_, 1);
    A2 = load_a_frag(Abase + 32 * C_ + kk + 32, C_, 1);
    A3 = load_a_frag(Abase + 48 * C_ + kk + 32, C_, 1);
  }
  acc[0] = wmma_bf16(A0, B0, acc[0]);
  acc[1] = wmma_bf16(A1, B0, acc[1]);
  acc[2] = wmma_bf16(A2, B0, acc[2]);
  acc[3] = wmma_bf16(A3, B0, acc[3]);

  const int seg = mt % 3;                  // 0=q, 1=k, 2=v (wave-uniform)
  const int hh  = mt / 3;
  const long bhd = (long)b * HEADS_ + hh;
  const int lcol = nt * 16 + n;
  if (seg == 0) {
#pragma unroll
    for (int i = 0; i < 4; ++i)
#pragma unroll
      for (int r = 0; r < 8; ++r) {
        const int cl = i * 16 + mb + r;
        qt[(bhd * L_ + lcol) * CH_ + cl] = (bf16)(acc[i][r] + bqkv[hh * 192 + cl]);
      }
  } else if (seg == 1) {
#pragma unroll
    for (int i = 0; i < 4; ++i)
#pragma unroll
      for (int r = 0; r < 8; ++r) {
        const int cl = i * 16 + mb + r;
        kt[(bhd * L_ + lcol) * CH_ + cl] = (bf16)(acc[i][r] + bqkv[hh * 192 + 64 + cl]);
      }
  } else {
#pragma unroll
    for (int i = 0; i < 4; ++i)
#pragma unroll
      for (int r = 0; r < 8; ++r) {
        const int cl = i * 16 + mb + r;
        vb[(bhd * CH_ + cl) * L_ + lcol] = (bf16)(acc[i][r] + bqkv[hh * 192 + 128 + cl]);
      }
  }
}

// ---------------- Flash attention ----------------
// One block (8 waves) per (head, 128-query super-tile). K/V tiles for each 32-key step are
// cooperatively staged into double-buffered LDS (8 KB/step) and shared by all 8 waves:
// 8x less L2 traffic, one __syncthreads per step, staging overlapped with compute.
__global__ void attn_kernel(const bf16* __restrict__ qt, const bf16* __restrict__ kt,
                            const bf16* __restrict__ vb, bf16* __restrict__ at) {
  __shared__ __align__(16) bf16 Ks[2][32 * CH_];   // [s(32)][c(64)]
  __shared__ __align__(16) bf16 Vs[2][CH_ * 32];   // [c(64)][s(32)]
  __shared__ __align__(16) bf16 Pb[8][16 * 32];    // per-wave P staging

  const int tid  = threadIdx.x;
  const int widl = tid >> 5;
  const int bh   = blockIdx.x >> 5;                // 16 head-batches
  const int bt   = blockIdx.x & 31;                // 32 query super-tiles
  const int t0   = bt * 128 + widl * 16;
  const int l    = lane_id();
  const int n    = l & 15;
  const int mb   = (l & 16) >> 1;

  const bf16* qtp = qt + (long)bh * L_ * CH_;
  const bf16* ktp = kt + (long)bh * L_ * CH_;
  const bf16* vbp = vb + (long)bh * CH_ * L_;

  // stage helper indices: K tile is one contiguous 4 KB block; V tile is 64 rows x 64 B
  const int vc = tid >> 2;
  const int vj = (tid & 3) << 3;

  // stage step 0 into buffer 0
  ((uint4*)&Ks[0][0])[tid] = ((const uint4*)ktp)[tid];
  *(uint4*)&Vs[0][vc * 32 + vj] = *(const uint4*)(vbp + (long)vc * L_ + vj);

  // Q fragments (loop-invariant, contiguous per lane)
  const v16bf qa0 = load_a_frag(qtp + (long)t0 * CH_,      CH_, 1);
  const v16bf qa1 = load_a_frag(qtp + (long)t0 * CH_ + 32, CH_, 1);

  v8f o0 = zero_v8f(), o1 = zero_v8f(), o2 = zero_v8f(), o3 = zero_v8f();
  float mrow[8], lrow[8];
#pragma unroll
  for (int r = 0; r < 8; ++r) { mrow[r] = -3.0e38f; lrow[r] = 0.0f; }

  bf16* Pw = &Pb[widl][0];

  for (int s0 = 0; s0 < L_; s0 += 32) {
    const int buf = (s0 >> 5) & 1;
    __syncthreads();
    if (s0 + 32 < L_) {                  // stage next step into the other buffer
      const int nb = buf ^ 1;
      ((uint4*)&Ks[nb][0])[tid] = ((const uint4*)(ktp + (long)(s0 + 32) * CH_))[tid];
      *(uint4*)&Vs[nb][vc * 32 + vj] = *(const uint4*)(vbp + (long)vc * L_ + (s0 + 32) + vj);
    }
    if (s0 + 64 < L_)                    // warm L2 two steps ahead (global_prefetch_b8)
      __builtin_prefetch(ktp + (long)(s0 + 64) * CH_ + (l << 2), 0, 0);

    const bf16* Kb = &Ks[buf][0];
    const bf16* Vb = &Vs[buf][0];

    // score tiles S0: keys [s0,s0+16), S1: [s0+16,s0+32); from LDS (contiguous per lane)
    v8f sa = zero_v8f(), sb = zero_v8f();
    sa = wmma_bf16(qa0, load_b_frag(Kb,                 1, CH_), sa);
    sa = wmma_bf16(qa1, load_b_frag(Kb + 32,            1, CH_), sa);
    sb = wmma_bf16(qa0, load_b_frag(Kb + 16 * CH_,      1, CH_), sb);
    sb = wmma_bf16(qa1, load_b_frag(Kb + 16 * CH_ + 32, 1, CH_), sb);

#pragma unroll
    for (int r = 0; r < 8; ++r) {
      float p0 = sa[r] * 0.125f;         // scale^2 = 1/sqrt(64)
      float p1 = sb[r] * 0.125f;
      float mx = fmaxf(p0, p1);
#pragma unroll
      for (int off = 8; off > 0; off >>= 1) mx = fmaxf(mx, __shfl_xor(mx, off, 32));
      const float mn    = fmaxf(mrow[r], mx);
      const float alpha = __expf(mrow[r] - mn);
      mrow[r] = mn;
      p0 = __expf(p0 - mn);
      p1 = __expf(p1 - mn);
      float ps = p0 + p1;
#pragma unroll
      for (int off = 8; off > 0; off >>= 1) ps += __shfl_xor(ps, off, 32);
      lrow[r] = lrow[r] * alpha + ps;
      o0[r] *= alpha; o1[r] *= alpha; o2[r] *= alpha; o3[r] *= alpha;
      Pw[(mb + r) * 32 + n]      = (bf16)p0;
      Pw[(mb + r) * 32 + 16 + n] = (bf16)p1;
    }
    const v16bf pa = load_a_frag(Pw, 32, 1);
    // O[t,c] += P[t,s] * V^T[s,c]; B element (k=s, n=c) at Vs[(ct*16+n)*32 + k]
    o0 = wmma_bf16(pa, load_b_frag(Vb,           1, 32), o0);
    o1 = wmma_bf16(pa, load_b_frag(Vb + 16 * 32, 1, 32), o1);
    o2 = wmma_bf16(pa, load_b_frag(Vb + 32 * 32, 1, 32), o2);
    o3 = wmma_bf16(pa, load_b_frag(Vb + 48 * 32, 1, 32), o3);
  }

  // normalize and write TRANSPOSED: at[b, t, h*64 + c] (c contiguous for proj B-fragments)
  const int b = bh >> 3, h = bh & 7;
#pragma unroll
  for (int r = 0; r < 8; ++r) {
    const float inv = 1.0f / lrow[r];
    const long  row = (long)b * L_ + (t0 + mb + r);
    at[row * C_ + h * CH_ +  0 + n] = (bf16)(o0[r] * inv);
    at[row * C_ + h * CH_ + 16 + n] = (bf16)(o1[r] * inv);
    at[row * C_ + h * CH_ + 32 + n] = (bf16)(o2[r] * inv);
    at[row * C_ + h * CH_ + 48 + n] = (bf16)(o3[r] * inv);
  }
}

// ---------------- proj GEMM + bias + residual (fp32 out, pipelined k-loop) ----------------
__global__ void gemm_proj_kernel(const bf16* __restrict__ wp, const bf16* __restrict__ at,
                                 const float* __restrict__ bp, const float* __restrict__ x,
                                 float* __restrict__ out) {
  const int wid = blockIdx.x * (blockDim.x >> 5) + (threadIdx.x >> 5);
  const int NT = L_ / 16;        // 256
  const int MT = C_ / 64;        // 8
  const int nt = wid % NT;
  const int mt = (wid / NT) % MT;
  const int b  = wid / (NT * MT);
  const int l  = lane_id();
  const int n  = l & 15;
  const int mb = (l & 16) >> 1;

  v8f acc[4];
#pragma unroll
  for (int i = 0; i < 4; ++i) acc[i] = zero_v8f();

  const bf16* Bbase = at + ((long)b * L_ + nt * 16) * C_;
  const bf16* Abase = wp + (long)mt * 64 * C_;

  v16bf B0 = load_b_frag(Bbase, 1, C_);
  v16bf A0 = load_a_frag(Abase,           C_, 1);
  v16bf A1 = load_a_frag(Abase + 16 * C_, C_, 1);
  v16bf A2 = load_a_frag(Abase + 32 * C_, C_, 1);
  v16bf A3 = load_a_frag(Abase + 48 * C_, C_, 1);
  for (int kk = 0; kk < C_ - 32; kk += 32) {
    acc[0] = wmma_bf16(A0, B0, acc[0]);
    acc[1] = wmma_bf16(A1, B0, acc[1]);
    acc[2] = wmma_bf16(A2, B0, acc[2]);
    acc[3] = wmma_bf16(A3, B0, acc[3]);
    B0 = load_b_frag(Bbase + kk + 32, 1, C_);
    A0 = load_a_frag(Abase + kk + 32,           C_, 1);
    A1 = load_a_frag(Abase + 16 * C_ + kk + 32, C_, 1);
    A2 = load_a_frag(Abase + 32 * C_ + kk + 32, C_, 1);
    A3 = load_a_frag(Abase + 48 * C_ + kk + 32, C_, 1);
  }
  acc[0] = wmma_bf16(A0, B0, acc[0]);
  acc[1] = wmma_bf16(A1, B0, acc[1]);
  acc[2] = wmma_bf16(A2, B0, acc[2]);
  acc[3] = wmma_bf16(A3, B0, acc[3]);

#pragma unroll
  for (int i = 0; i < 4; ++i) {
#pragma unroll
    for (int r = 0; r < 8; ++r) {
      const int  o   = mt * 64 + i * 16 + mb + r;
      const long idx = (long)b * C_ * L_ + (long)o * L_ + nt * 16 + n;
      out[idx] = acc[i][r] + bp[o] + x[idx];
    }
  }
}

extern "C" void kernel_launch(void* const* d_in, const int* in_sizes, int n_in,
                              void* d_out, int out_size, void* d_ws, size_t ws_size,
                              hipStream_t stream) {
  const float* x      = (const float*)d_in[0];
  const float* gamma  = (const float*)d_in[1];
  const float* beta   = (const float*)d_in[2];
  const float* w_qkv  = (const float*)d_in[3];
  const float* b_qkv  = (const float*)d_in[4];
  const float* w_proj = (const float*)d_in[5];
  const float* b_proj = (const float*)d_in[6];
  float* out = (float*)d_out;

  char* ws = (char*)d_ws;
  float* stats = (float*)ws;                          //  512 B (+pad to 1 KiB)
  bf16*  wqb   = (bf16*)(ws + 1024);                  //  1.5 MiB
  bf16*  wpb   = (bf16*)(ws + 1024 + 1572864);        //  0.5 MiB
  bf16*  xnt   = (bf16*)(ws + 2098176);               //  8 MiB  xn  [b, l, c]
  bf16*  qt    = (bf16*)(ws + 10486784);              //  8 MiB  q   [bh, t, c]
  bf16*  kt    = (bf16*)(ws + 18875392);              //  8 MiB  k   [bh, s, c]
  bf16*  vb    = (bf16*)(ws + 27264000);              //  8 MiB  v   [bh, c, s]
  bf16*  at    = (bf16*)(ws + 35652608);              //  8 MiB  a   [b, l, c]   (end ~42 MiB)

  gn_stats_kernel<<<B_ * GROUPS_, 256, 0, stream>>>(x, stats);
  cast_bf16_kernel<<<768, 256, 0, stream>>>(w_qkv, wqb, 3 * C_ * C_);
  cast_bf16_kernel<<<256, 256, 0, stream>>>(w_proj, wpb, C_ * C_);
  gn_apply_kernel<<<4096, 256, 0, stream>>>(x, gamma, beta, stats, xnt);
  gemm_qkv_kernel<<<1536, 256, 0, stream>>>(wqb, xnt, b_qkv, qt, kt, vb);  // 12288 waves
  attn_kernel<<<512, 256, 0, stream>>>(qt, kt, vb, at);                    // 4096 waves
  gemm_proj_kernel<<<512, 256, 0, stream>>>(wpb, at, b_proj, x, out);      // 4096 waves
}
